// PhysNetCore_53395033424547
// MI455X (gfx1250) — matrix operand order repaired and checked by reading, not compiled, not run on hardware
//
#include <hip/hip_runtime.h>
#include <hip/hip_bf16.h>

typedef _Float16 v16h __attribute__((ext_vector_type(16)));
typedef _Float16 v8h  __attribute__((ext_vector_type(8)));
typedef float    v8f  __attribute__((ext_vector_type(8)));
typedef float    v4f  __attribute__((ext_vector_type(4)));

__device__ __forceinline__ float softplus_f(float x) {
    // numerically stable log1p(exp(x))
    return fmaxf(x, 0.0f) + log1pf(__expf(-fabsf(x)));
}

// ---------------------------------------------------------------------------
// Transpose + fp16-convert a 128x128 f32 weight matrix: Wt[n*128+k] = W[k*128+n]
// ---------------------------------------------------------------------------
__global__ __launch_bounds__(256) void transposeW_kernel(
    const float* __restrict__ W, _Float16* __restrict__ Wt)
{
    int idx = blockIdx.x * 256 + threadIdx.x;   // 0..16383
    int n = idx >> 7;
    int k = idx & 127;
    Wt[n * 128 + k] = (_Float16)W[k * 128 + n];
}

// ---------------------------------------------------------------------------
// Fused WMMA GEMM: out[N,128] = POST( PRE(A)[N,128] @ W[128,128] + bias )
// PRE:  0 = identity, 1 = softplus
// POST: 0 = softplus(softplus(acc+b))      (embedding projections)
//       1 = softplus(acc+b)                (residual hidden)
//       2 = aux[row,col] + acc + b         (residual output)
//       3 = acc + b + gate[col]*aux[row,col]  ("updated")
// Wt is the f16 transposed weight: Wt[n*128 + k] = W[k][n]
// ---------------------------------------------------------------------------
template<int PRE, int POST>
__global__ __launch_bounds__(256) void gemm128_kernel(
    const float* __restrict__ A,
    const _Float16* __restrict__ Wt,
    const float* __restrict__ bias,
    const float* __restrict__ aux,
    const float* __restrict__ gate,
    float* __restrict__ out,
    int N)
{
    __shared__ _Float16 As[128 * 128];   // 32 KB A tile in f16
    const int tid    = threadIdx.x;
    const int m_base = blockIdx.x * 128;

    // ---- stage A tile: f32 -> (PRE) -> f16, row-major in LDS ----
    for (int i = tid; i < 128 * 32; i += 256) {   // float4 granularity
        int r  = i >> 5;        // row in tile
        int c4 = i & 31;        // float4 index in row
        int row = m_base + r;
        v4f v = {0.f, 0.f, 0.f, 0.f};
        if (row < N) v = *(const v4f*)(A + (size_t)row * 128 + c4 * 4);
        if (PRE == 1) {
            v.x = softplus_f(v.x); v.y = softplus_f(v.y);
            v.z = softplus_f(v.z); v.w = softplus_f(v.w);
        }
        _Float16* dst = &As[r * 128 + c4 * 4];
        dst[0] = (_Float16)v.x; dst[1] = (_Float16)v.y;
        dst[2] = (_Float16)v.z; dst[3] = (_Float16)v.w;
    }
    __syncthreads();

    const int wave = tid >> 5;        // 0..7 -> 16-row strip
    const int lane = tid & 31;
    const int m0   = wave * 16;
    const int l16  = lane & 15;
    const int hi   = lane >> 4;       // 0 or 1

    v8f acc[8];
    #pragma unroll
    for (int n = 0; n < 8; ++n) acc[n] = (v8f){0,0,0,0,0,0,0,0};

    #pragma unroll
    for (int kk = 0; kk < 4; ++kk) {
        const int k0 = kk * 32;
        // A fragment (16x32 f16): lanes 0-15 hold K {0..15}, lanes 16-31 K {16..31}
        const int c0 = k0 + hi * 16;
        v8h alo = *(const v8h*)(&As[(m0 + l16) * 128 + c0]);
        v8h ahi = *(const v8h*)(&As[(m0 + l16) * 128 + c0 + 8]);
        v16h a = __builtin_shufflevector(alo, ahi,
                                         0,1,2,3,4,5,6,7,8,9,10,11,12,13,14,15);
        #pragma unroll
        for (int n = 0; n < 8; ++n) {
            // B fragment (32x16 f16): lane -> column n*16+l16, contiguous 16 K values
            const _Float16* bp = Wt + (size_t)(n * 16 + l16) * 128 + k0 + hi * 16;
            v16h b = *(const v16h*)bp;
            acc[n] = __builtin_amdgcn_wmma_f32_16x16x32_f16(
                false, a, false, b, (short)0, acc[n], false, false);
        }
    }

    // ---- epilogue: C layout: vgpr r -> row m0 + r + hi*8, col n*16 + l16 ----
    #pragma unroll
    for (int n = 0; n < 8; ++n) {
        const int col = n * 16 + l16;
        const float b = bias[col];
        #pragma unroll
        for (int r = 0; r < 8; ++r) {
            int row = m_base + m0 + r + hi * 8;
            if (row < N) {
                float v = acc[n][r] + b;
                if (POST == 0)      v = softplus_f(softplus_f(v));
                else if (POST == 1) v = softplus_f(v);
                else if (POST == 2) v = aux[(size_t)row * 128 + col] + v;
                else if (POST == 3) v = v + gate[col] * aux[(size_t)row * 128 + col];
                out[(size_t)row * 128 + col] = v;
            }
        }
    }
}

// ---------------------------------------------------------------------------
// Edge kernel: xi[idx_i[p], f] += (f_ij[p,:] @ Wg)[f] * xjf[idx_j[p], f]
// ---------------------------------------------------------------------------
__global__ __launch_bounds__(256) void edge_kernel(
    const float* __restrict__ f_ij,
    const int*   __restrict__ idx_i,
    const int*   __restrict__ idx_j,
    const float* __restrict__ Wg,     // [16,128]
    const float* __restrict__ xjf,    // [N,128]
    float*       __restrict__ xi,     // [N,128]  (accumulated into)
    int P)
{
    __shared__ float wg[16 * 128];    // 8 KB
    for (int i = threadIdx.x; i < 2048; i += 256) wg[i] = Wg[i];
    __syncthreads();

    const int total  = P * 128;                 // 102.4M < 2^31
    const int stride = gridDim.x * 256;
    for (int idx = blockIdx.x * 256 + threadIdx.x; idx < total; idx += stride) {
        int p = idx >> 7;
        int f = idx & 127;
        const float* fr = f_ij + p * 16;
        float g = 0.f;
        #pragma unroll
        for (int r = 0; r < 16; ++r) g += fr[r] * wg[r * 128 + f];
        int j = idx_j[p];
        int i = idx_i[p];
        float val = g * xjf[(size_t)j * 128 + f];
        unsafeAtomicAdd(xi + (size_t)i * 128 + f, val);
    }
}

// ---------------------------------------------------------------------------
// Output projection: pred[N,2] = y[N,128] @ Wout[128,2] + bout
// ---------------------------------------------------------------------------
__global__ __launch_bounds__(256) void outproj_kernel(
    const float* __restrict__ y,
    const float* __restrict__ Wout,
    const float* __restrict__ bout,
    float* __restrict__ pred,
    int N)
{
    __shared__ float w[256];
    w[threadIdx.x] = Wout[threadIdx.x];
    __syncthreads();

    int row = blockIdx.x * 256 + threadIdx.x;
    if (row >= N) return;
    float a0 = bout[0], a1 = bout[1];
    const v4f* yr = (const v4f*)(y + (size_t)row * 128);
    #pragma unroll 8
    for (int k4 = 0; k4 < 32; ++k4) {
        v4f v = yr[k4];
        int k = k4 * 4;
        a0 += v.x * w[(k+0)*2+0] + v.y * w[(k+1)*2+0]
            + v.z * w[(k+2)*2+0] + v.w * w[(k+3)*2+0];
        a1 += v.x * w[(k+0)*2+1] + v.y * w[(k+1)*2+1]
            + v.z * w[(k+2)*2+1] + v.w * w[(k+3)*2+1];
    }
    pred[row * 2 + 0] = a0;
    pred[row * 2 + 1] = a1;
}

// ---------------------------------------------------------------------------
extern "C" void kernel_launch(void* const* d_in, const int* in_sizes, int n_in,
                              void* d_out, int out_size, void* d_ws, size_t ws_size,
                              hipStream_t stream)
{
    const float* emb     = (const float*)d_in[0];
    const float* f_ij    = (const float*)d_in[1];
    const int*   idx_i   = (const int*)  d_in[2];
    const int*   idx_j   = (const int*)  d_in[3];
    const float* Wi      = (const float*)d_in[4];
    const float* bi      = (const float*)d_in[5];
    const float* Wj      = (const float*)d_in[6];
    const float* bj      = (const float*)d_in[7];
    const float* Wg      = (const float*)d_in[8];
    const float* Wv      = (const float*)d_in[9];
    const float* bv      = (const float*)d_in[10];
    const float* gate    = (const float*)d_in[11];
    const float* res_w1  = (const float*)d_in[12];
    const float* res_b1  = (const float*)d_in[13];
    const float* res_w2  = (const float*)d_in[14];
    const float* res_b2  = (const float*)d_in[15];
    const float* ores_w1 = (const float*)d_in[16];
    const float* ores_b1 = (const float*)d_in[17];
    const float* ores_w2 = (const float*)d_in[18];
    const float* ores_b2 = (const float*)d_in[19];
    const float* Wout    = (const float*)d_in[20];
    const float* bout    = (const float*)d_in[21];

    const int N = in_sizes[0] / 128;
    const int P = in_sizes[2];
    const int NRES = in_sizes[12] / (128 * 128);   // 3

    // workspace layout
    char* ws = (char*)d_ws;
    _Float16* wt = (_Float16*)ws;                  // 11 matrices of 16384 halfs
    const size_t WT_BYTES = (size_t)11 * 16384 * sizeof(_Float16);  // 360448, 256B-mult
    float* xi  = (float*)(ws + WT_BYTES);
    float* xjf = xi + (size_t)N * 128;
    float* h   = xjf;                              // alias: xjf dead after edge kernel

    float* pred    = (float*)d_out;
    float* updated = (float*)d_out + (size_t)N * 2;
    float* y       = xi;                           // alias: xi dead after "updated"

    // weight slots: 0=Wi 1=Wj 2=Wv 3..5=res_w1 6..8=res_w2 9=ores_w1 10=ores_w2
    transposeW_kernel<<<64, 256, 0, stream>>>(Wi, wt + 0 * 16384);
    transposeW_kernel<<<64, 256, 0, stream>>>(Wj, wt + 1 * 16384);
    transposeW_kernel<<<64, 256, 0, stream>>>(Wv, wt + 2 * 16384);
    for (int k = 0; k < NRES; ++k) {
        transposeW_kernel<<<64, 256, 0, stream>>>(res_w1 + (size_t)k * 16384, wt + (3 + k) * 16384);
        transposeW_kernel<<<64, 256, 0, stream>>>(res_w2 + (size_t)k * 16384, wt + (6 + k) * 16384);
    }
    transposeW_kernel<<<64, 256, 0, stream>>>(ores_w1, wt + 9 * 16384);
    transposeW_kernel<<<64, 256, 0, stream>>>(ores_w2, wt + 10 * 16384);

    const int gb = (N + 127) / 128;

    // xi = act(act(emb @ Wi + bi)); xjf = act(act(emb @ Wj + bj))
    gemm128_kernel<0, 0><<<gb, 256, 0, stream>>>(emb, wt + 0 * 16384, bi, nullptr, nullptr, xi,  N);
    gemm128_kernel<0, 0><<<gb, 256, 0, stream>>>(emb, wt + 1 * 16384, bj, nullptr, nullptr, xjf, N);

    // xi[idx_i] += (f_ij @ Wg) * xjf[idx_j]
    edge_kernel<<<4096, 256, 0, stream>>>(f_ij, idx_i, idx_j, Wg, xjf, xi, P);

    // 3 residual blocks on xi (h aliases xjf, now free)
    for (int k = 0; k < NRES; ++k) {
        gemm128_kernel<1, 1><<<gb, 256, 0, stream>>>(xi, wt + (3 + k) * 16384,
                                                     res_b1 + (size_t)k * 128,
                                                     nullptr, nullptr, h, N);
        gemm128_kernel<0, 2><<<gb, 256, 0, stream>>>(h, wt + (6 + k) * 16384,
                                                     res_b2 + (size_t)k * 128,
                                                     xi, nullptr, xi, N);
    }

    // updated = gate*emb + act(xi) @ Wv + bv   (written straight into d_out slot)
    gemm128_kernel<1, 3><<<gb, 256, 0, stream>>>(xi, wt + 2 * 16384, bv, emb, gate, updated, N);

    // output residual block: y = updated + act(act(updated)@w1+b1)@w2+b2
    gemm128_kernel<1, 1><<<gb, 256, 0, stream>>>(updated, wt + 9 * 16384, ores_b1,
                                                 nullptr, nullptr, h, N);
    gemm128_kernel<0, 2><<<gb, 256, 0, stream>>>(h, wt + 10 * 16384, ores_b2,
                                                 updated, nullptr, y, N);

    // prediction = y @ Wout + bout
    outproj_kernel<<<(N + 255) / 256, 256, 0, stream>>>(y, Wout, bout, pred, N);
}
